// RefOnce_34522947125794
// MI455X (gfx1250) — matrix operand order repaired and checked by reading, not compiled, not run on hardware
//
#include <hip/hip_runtime.h>
#include <hip/hip_bf16.h>

// ---------------------------------------------------------------------------
// Problem constants
// ---------------------------------------------------------------------------
#define BS   32
#define C    64
#define H    96
#define W    96
#define T    64
#define HP   98          // padded H
#define WP   98          // padded W
#define K576 576         // C*9
#define KC   18          // K576 / 32
#define HW   (H*W)       // 9216

#define XPAD_ELEMS ((size_t)BS * C * HP * WP)          // 19,668,992 bf16 elems
#define XPAD_BYTES (XPAD_ELEMS * 2)
#define WFRAG_ELEMS (KC * 4 * 32 * 16)                 // 36,864 bf16 elems

typedef __attribute__((ext_vector_type(16))) __bf16 v16bf;
typedef __attribute__((ext_vector_type(8)))  float  v8f;

// f32 -> bf16 round-to-nearest-even
__device__ __forceinline__ unsigned short f2bf(float f) {
    unsigned int u = __float_as_uint(f);
    unsigned int r = u + 0x7FFFu + ((u >> 16) & 1u);
    return (unsigned short)(r >> 16);
}

// ---------------------------------------------------------------------------
// Utility: grid-stride zero of u32 words
// ---------------------------------------------------------------------------
__global__ void zero_u32_kernel(unsigned int* __restrict__ p, size_t n) {
    for (size_t i = (size_t)blockIdx.x * blockDim.x + threadIdx.x; i < n;
         i += (size_t)gridDim.x * blockDim.x)
        p[i] = 0u;
}

// ---------------------------------------------------------------------------
// Convert x [BS][C][H][W] f32 -> bf16 into zero-padded [BS][C][HP][WP]
// (borders pre-zeroed by zero kernel -> conv needs no bounds checks)
// ---------------------------------------------------------------------------
__global__ void cvt_x_kernel(const float* __restrict__ x,
                             unsigned short* __restrict__ xpad) {
    const size_t n = (size_t)BS * C * H * W;
    for (size_t i = (size_t)blockIdx.x * blockDim.x + threadIdx.x; i < n;
         i += (size_t)gridDim.x * blockDim.x) {
        int w  = (int)(i % W);
        size_t t1 = i / W;
        int h  = (int)(t1 % H);
        size_t bc = t1 / H;                       // b*C + c
        xpad[bc * (HP * WP) + (size_t)(h + 1) * WP + (w + 1)] = f2bf(x[i]);
    }
}

// ---------------------------------------------------------------------------
// Pack conv weights [O=64][I=64][3][3] f32 into bf16 A-fragment order.
// GEMM K ordering is (kh,kw)-major, ic-minor:  k = (kh*3+kw)*64 + ic
// so that B fragments are contiguous channel runs in the LDS halo tile.
//   wfrag[kc][mtile][lane][e]  (flat: kc*2048 + mtile*512 + lane*16 + e)
// A 16x32 lane layout used here (contiguous K, same order as B):
//   lane<16 -> M=lane,    K=e
//   lane>=16-> M=lane-16, K=e+16
// ---------------------------------------------------------------------------
__global__ void pack_w_kernel(const float* __restrict__ wsrc,
                              unsigned short* __restrict__ wfrag) {
    int i = blockIdx.x * blockDim.x + threadIdx.x;
    if (i >= WFRAG_ELEMS) return;
    int e    = i & 15;
    int lane = (i >> 4) & 31;
    int mt   = (i >> 9) & 3;
    int kc   = i >> 11;
    bool lanelo = lane < 16;
    int M  = lane & 15;
    int Kl = e + (lanelo ? 0 : 16);
    int o  = mt * 16 + M;
    int k  = kc * 32 + Kl;          // k = (kh*3+kw)*64 + ic
    int ic   = k & 63;
    int khkw = k >> 6;
    int kh   = khkw / 3;
    int kw   = khkw - kh * 3;
    wfrag[i] = f2bf(wsrc[(((size_t)o * C + ic) * 3 + kh) * 3 + kw]);
}

// ---------------------------------------------------------------------------
// Implicit-GEMM conv3x3 + BN + ReLU with v_wmma_f32_16x16x32_bf16.
// Block = 256 threads (8 wave32), tile = 64 out-ch x (8 rows x 16 cols).
// Wave w owns pixel row w; accumulates 4 MxN=16x16 tiles over K=576.
// LDS halo tile stored CHANNEL-MINOR: lds_x[row][col][ch], so a lane's
// 16 consecutive-K B-fragment elements are one aligned 32-B vector
// (2 x ds_load_b128 instead of 16 x ds_load_u16 + packing).
// FUSE_POOL=false: write bf16 into next padded activation buffer.
// FUSE_POOL=true : global-avg-pool reduction (LDS ds_add_f32 + 1 global
//                  atomic per channel per block) -- conv2 output never
//                  written to memory.
// ---------------------------------------------------------------------------
template <bool FUSE_POOL>
__global__ __launch_bounds__(256)
void conv3x3_bn_relu_wmma(const unsigned short* __restrict__ xpad,   // [BS][C][HP][WP] bf16
                          const unsigned short* __restrict__ wfrag,  // packed A fragments
                          const float* __restrict__ bn_g,
                          const float* __restrict__ bn_b,
                          const float* __restrict__ bn_m,
                          const float* __restrict__ bn_v,
                          unsigned short* __restrict__ ypad,         // !FUSE_POOL
                          float* __restrict__ pool_acc)              // FUSE_POOL
{
    // [10 rows][18 cols][64 ch], channel-minor; 23,040 B
    __shared__ __align__(32) unsigned short lds_x[10 * 18 * C];
    __shared__ float lds_pool[C];

    const int bx   = blockIdx.x;        // 0..5   (W/16 tiles)
    const int by   = blockIdx.y;        // 0..11  (H/8 tiles)
    const int b    = blockIdx.z;        // batch
    const int tid  = threadIdx.x;
    const int lane = tid & 31;
    const int wave = tid >> 5;
    const int h0   = by * 8;
    const int w0   = bx * 16;

    // ---- stage halo tile: rows h0..h0+9, cols w0..w0+17, all 64 channels.
    // Global reads coalesced (consecutive tid -> consecutive cols, fixed ch);
    // transpose happens on the LDS-write side.
    const unsigned short* xb = xpad + (size_t)b * C * HP * WP;
    for (int idx = tid; idx < C * 10 * 18; idx += 256) {
        int ch  = idx / 180;
        int rem = idx - ch * 180;
        int r   = rem / 18;
        int cc  = rem - r * 18;
        lds_x[(r * 18 + cc) * C + ch] =
            xb[(size_t)ch * (HP * WP) + (size_t)(h0 + r) * WP + (w0 + cc)];
    }
    if constexpr (FUSE_POOL) {
        if (tid < C) lds_pool[tid] = 0.0f;
    }
    __syncthreads();

    const int  row    = wave;        // pixel row in tile (0..7)
    const int  col    = lane & 15;   // pixel col in tile (0..15) == N
    const bool lanelo = lane < 16;

    v8f acc0 = {}, acc1 = {}, acc2 = {}, acc3 = {};

    for (int kc = 0; kc < KC; ++kc) {
        // ---- B fragment (32x16 bf16, K x N). K ordering (kh,kw,ic):
        // chunk kc covers (kh,kw) = kc/2, ic in [ (kc&1)*32, +32 ).
        // lane<16 -> K=kk..kk+15, lane>=16 -> K=kk+16..kk+31 => 16
        // consecutive channels at one halo pixel = one 32-B LDS vector.
        const int khkw = kc >> 1;
        const int kh   = khkw / 3;
        const int kw   = khkw - kh * 3;
        const int icb  = ((kc & 1) << 5) + (lanelo ? 0 : 16);
        const v16bf bfrag =
            *(const v16bf*)&lds_x[((row + kh) * 18 + (col + kw)) * C + icb];

        // ---- A fragments: contiguous 32-B vectors per lane, L2-hot
        const unsigned short* wk = wfrag + ((size_t)(kc * 4) * 32 + lane) * 16;
        v16bf a0 = *(const v16bf*)(wk);
        v16bf a1 = *(const v16bf*)(wk + 512);
        v16bf a2 = *(const v16bf*)(wk + 1024);
        v16bf a3 = *(const v16bf*)(wk + 1536);

        acc0 = __builtin_amdgcn_wmma_f32_16x16x32_bf16(false, a0, false, bfrag,
                                                       (short)0, acc0, false, false);
        acc1 = __builtin_amdgcn_wmma_f32_16x16x32_bf16(false, a1, false, bfrag,
                                                       (short)0, acc1, false, false);
        acc2 = __builtin_amdgcn_wmma_f32_16x16x32_bf16(false, a2, false, bfrag,
                                                       (short)0, acc2, false, false);
        acc3 = __builtin_amdgcn_wmma_f32_16x16x32_bf16(false, a3, false, bfrag,
                                                       (short)0, acc3, false, false);
    }

    // ---- epilogue: BN + ReLU, then either store bf16 padded or pool-reduce.
    // C/D layout: VGPR r holds M = r (lanes 0-15) / r+8 (lanes 16-31), N = lane%16.
    const int moff = lanelo ? 0 : 8;
    auto emit = [&](const v8f& acc, int mtile) {
#pragma unroll
        for (int r = 0; r < 8; ++r) {
            int o = mtile * 16 + r + moff;
            float s  = bn_g[o] * rsqrtf(bn_v[o] + 1e-5f);
            float tb = bn_b[o] - bn_m[o] * s;
            float val = fmaf(acc[r], s, tb);
            val = val > 0.0f ? val : 0.0f;
            if constexpr (FUSE_POOL) {
                atomicAdd(&lds_pool[o], val);
            } else {
                ypad[(size_t)b * C * HP * WP + (size_t)o * HP * WP +
                     (size_t)(1 + h0 + row) * WP + (1 + w0 + col)] = f2bf(val);
            }
        }
    };
    emit(acc0, 0); emit(acc1, 1); emit(acc2, 2); emit(acc3, 3);

    if constexpr (FUSE_POOL) {
        __syncthreads();
        if (tid < C) atomicAdd(&pool_acc[b * C + tid], lds_pool[tid]);
    }
}

// ---------------------------------------------------------------------------
// Head: x_feat = pool/9216 -> MLP(64->128 relu ->64) -> softmax weights + CE.
// One block per sample, 128 threads. Tiny (0.5 MFLOP total) -> plain VALU.
// ---------------------------------------------------------------------------
__global__ __launch_bounds__(128)
void head_kernel(const float* __restrict__ pool_acc,
                 const float* __restrict__ w1, const float* __restrict__ b1,
                 const float* __restrict__ w2, const float* __restrict__ b2,
                 const int*   __restrict__ ref_types,
                 float* __restrict__ type_weights,
                 float* __restrict__ loss_out)
{
    __shared__ float sx[C];
    __shared__ float sh[2 * C];
    __shared__ float sl[T];
    const int b = blockIdx.x;
    const int t = threadIdx.x;

    if (t < C) sx[t] = pool_acc[b * C + t] * (1.0f / (float)HW);
    __syncthreads();

    {   // hidden[t], t in 0..127
        float acc = b1[t];
        for (int i = 0; i < C; ++i) acc = fmaf(sx[i], w1[i * (2 * C) + t], acc);
        sh[t] = acc > 0.0f ? acc : 0.0f;
    }
    __syncthreads();

    if (t < T) {
        float acc = b2[t];
        for (int i = 0; i < 2 * C; ++i) acc = fmaf(sh[i], w2[i * T + t], acc);
        sl[t] = acc;
    }
    __syncthreads();

    if (t == 0) {
        float mx = sl[0];
        for (int i = 1; i < T; ++i) mx = fmaxf(mx, sl[i]);
        float sum = 0.0f;
        for (int i = 0; i < T; ++i) sum += __expf(sl[i] - mx);
        float inv = 1.0f / sum;
        for (int i = 0; i < T; ++i) type_weights[b * T + i] = __expf(sl[i] - mx) * inv;
        int y = ref_types[b];
        float nll = -(sl[y] - mx - __logf(sum));
        atomicAdd(loss_out, nll * (1.0f / (float)BS));
    }
}

// ---------------------------------------------------------------------------
// Closed-form sequential EMA scatter. Block = type t, thread = channel c.
// Iterate samples back-to-front: momentum power == strictly-after count.
// ---------------------------------------------------------------------------
__global__ void ema_kernel(const float* __restrict__ ref_x,
                           const int*   __restrict__ ref_types,
                           const float* __restrict__ ref_proj,
                           float* __restrict__ ref_new_out,
                           float* __restrict__ ref_new_ws)
{
    const int t = blockIdx.x;
    const int c = threadIdx.x;
    const float m = 0.99f;
    float p = 1.0f, acc = 0.0f;
    for (int i = BS - 1; i >= 0; --i) {
        if (ref_types[i] == t) {
            acc = fmaf((1.0f - m) * p, ref_x[i * C + c], acc);
            p *= m;
        }
    }
    float v = fmaf(p, ref_proj[t * C + c], acc);   // p == m^counts[t]
    ref_new_out[t * C + c] = v;
    ref_new_ws[t * C + c]  = v;
}

// ---------------------------------------------------------------------------
// out[b][c] = sum_t type_weights[b][t] * ref_new[t][c] + ref_x[b][c]
// ---------------------------------------------------------------------------
__global__ void final_kernel(const float* __restrict__ type_weights,
                             const float* __restrict__ ref_new,
                             const float* __restrict__ ref_x,
                             float* __restrict__ out)
{
    const int b = blockIdx.x;
    const int c = threadIdx.x;
    float acc = 0.0f;
    for (int t = 0; t < T; ++t)
        acc = fmaf(type_weights[b * T + t], ref_new[t * C + c], acc);
    out[b * C + c] = acc + ref_x[b * C + c];
}

// ---------------------------------------------------------------------------
// Launcher
// ---------------------------------------------------------------------------
extern "C" void kernel_launch(void* const* d_in, const int* in_sizes, int n_in,
                              void* d_out, int out_size, void* d_ws, size_t ws_size,
                              hipStream_t stream) {
    const float* x        = (const float*)d_in[0];
    const float* ref_x    = (const float*)d_in[1];
    const int*   ref_ty   = (const int*)  d_in[2];
    const float* conv1_w  = (const float*)d_in[3];
    const float* bn1_g    = (const float*)d_in[4];
    const float* bn1_b    = (const float*)d_in[5];
    const float* bn1_m    = (const float*)d_in[6];
    const float* bn1_v    = (const float*)d_in[7];
    const float* conv2_w  = (const float*)d_in[8];
    const float* bn2_g    = (const float*)d_in[9];
    const float* bn2_b    = (const float*)d_in[10];
    const float* bn2_m    = (const float*)d_in[11];
    const float* bn2_v    = (const float*)d_in[12];
    const float* mlp_w1   = (const float*)d_in[13];
    const float* mlp_b1   = (const float*)d_in[14];
    const float* mlp_w2   = (const float*)d_in[15];
    const float* mlp_b2   = (const float*)d_in[16];
    const float* ref_proj = (const float*)d_in[17];

    // workspace layout (all offsets 32-B aligned)
    char* ws = (char*)d_ws;
    unsigned short* xpad1 = (unsigned short*)ws;                       // bf16
    unsigned short* xpad2 = xpad1 + XPAD_ELEMS;                        // bf16
    unsigned short* wf1   = (unsigned short*)(ws + 2 * XPAD_BYTES);
    unsigned short* wf2   = wf1 + WFRAG_ELEMS;
    float* pool_acc       = (float*)(wf2 + WFRAG_ELEMS);               // [BS*C]
    float* type_w         = pool_acc + BS * C;                         // [BS*T]
    float* ref_new_ws     = type_w + BS * T;                           // [T*C]

    float* out       = (float*)d_out;      // [BS*C]
    float* loss      = out + BS * C;       // [1]
    float* ref_new_o = loss + 1;           // [T*C]

    // 1) zero padded activation buffers (borders = conv zero-padding),
    //    pool accumulator, loss slot
    zero_u32_kernel<<<2048, 256, 0, stream>>>((unsigned int*)xpad1,
                                              (2 * XPAD_BYTES) / 4);
    zero_u32_kernel<<<8, 256, 0, stream>>>((unsigned int*)pool_acc, BS * C);
    zero_u32_kernel<<<1, 32, 0, stream>>>((unsigned int*)loss, 1);

    // 2) convert x to bf16 padded; pack both weight tensors into A-fragment order
    cvt_x_kernel<<<4608, 256, 0, stream>>>(x, xpad1);
    pack_w_kernel<<<(WFRAG_ELEMS + 255) / 256, 256, 0, stream>>>(conv1_w, wf1);
    pack_w_kernel<<<(WFRAG_ELEMS + 255) / 256, 256, 0, stream>>>(conv2_w, wf2);

    // 3) two WMMA convs; conv2 fuses the global average pool
    dim3 cgrid(W / 16, H / 8, BS);     // (6, 12, 32)
    conv3x3_bn_relu_wmma<false><<<cgrid, 256, 0, stream>>>(
        xpad1, wf1, bn1_g, bn1_b, bn1_m, bn1_v, xpad2, nullptr);
    conv3x3_bn_relu_wmma<true><<<cgrid, 256, 0, stream>>>(
        xpad2, wf2, bn2_g, bn2_b, bn2_m, bn2_v, nullptr, pool_acc);

    // 4) head MLP + softmax + CE loss
    head_kernel<<<BS, 128, 0, stream>>>(pool_acc, mlp_w1, mlp_b1, mlp_w2, mlp_b2,
                                        ref_ty, type_w, loss);

    // 5) EMA buffer update (writes ref_new output + ws copy)
    ema_kernel<<<T, C, 0, stream>>>(ref_x, ref_ty, ref_proj, ref_new_o, ref_new_ws);

    // 6) weighted reference + residual
    final_kernel<<<BS, C, 0, stream>>>(type_w, ref_new_ws, ref_x, out);
}
